// GraphSearchPolicy_58145267253328
// MI455X (gfx1250) — compile-verified, matrix operand order checked by reading, not run on hardware
//
#include <hip/hip_runtime.h>
#include <hip/hip_bf16.h>
#include <math.h>

// ---------------------------------------------------------------------------
// Problem constants (from reference): B=1024, AN=256, D=256, HD=512, AD=512
// ---------------------------------------------------------------------------
#define BATCH 1024
#define AN    256
#define DIM   256
#define HD    512
#define AD    512          // 2*DIM
#define INDIM 1024         // 2*DIM + HD
#define HUGEF 1e31f
#define EPSF  1e-20f

typedef __attribute__((ext_vector_type(16))) __bf16 v16bf;
typedef __attribute__((ext_vector_type(8)))  __bf16 v8bf;
typedef __attribute__((ext_vector_type(8)))  float  v8f;

// ---------------------------------------------------------------------------
// Prep: f32 [R,C] row-major  ->  bf16 [C,R] row-major (transposed)
// ---------------------------------------------------------------------------
__global__ void k_transpose_bf16(const float* __restrict__ in,
                                 __bf16* __restrict__ out, int R, int C) {
  int idx = blockIdx.x * blockDim.x + threadIdx.x;
  if (idx >= R * C) return;
  int r = idx / C, c = idx - r * C;
  out[(size_t)c * R + r] = (__bf16)in[(size_t)r * C + c];
}

// ---------------------------------------------------------------------------
// Prep: Xin[b, 0:256]=ent_emb[e[b]], [256:768]=hstate[b], [768:1024]=rel_emb[q[b]]
// ---------------------------------------------------------------------------
__global__ void k_build_xin(const int* __restrict__ e, const int* __restrict__ q,
                            const float* __restrict__ hstate,
                            const float* __restrict__ ent_emb,
                            const float* __restrict__ rel_emb,
                            __bf16* __restrict__ xin) {
  int idx = blockIdx.x * blockDim.x + threadIdx.x;  // BATCH*INDIM threads
  int b = idx >> 10;          // / 1024
  int c = idx & 1023;
  float v;
  if (c < DIM)            v = ent_emb[(size_t)e[b] * DIM + c];
  else if (c < DIM + HD)  v = hstate[(size_t)b * HD + (c - DIM)];
  else                    v = rel_emb[(size_t)q[b] * DIM + (c - DIM - HD)];
  xin[idx] = (__bf16)v;
}

// ---------------------------------------------------------------------------
// WMMA GEMM:  out[M,N] = op(A[M,K] @ BT[N,K]^T + bias[N])
// A, BT in bf16 row-major; one wave computes a 16x64 tile (4 accumulators,
// A-fragment reused across 4 v_wmma_f32_16x16x32_bf16 per k-step).
// ISA fragment layout: lane = (half = lane>>4, ln = lane&15)
//   A frag  : elems 0..7 = A[m][kb + half*8 + j],  8..15 = A[m][kb+16+half*8+j]
//   B frag  : same pattern on BT row n (B stored transposed)
//   C/D f32 : elem i -> row (tile_m*16 + i + 8*half), col (tile_n*16 + ln)
// ---------------------------------------------------------------------------
__global__ void k_wmma_gemm(const __bf16* __restrict__ A,
                            const __bf16* __restrict__ BT,
                            const float* __restrict__ bias,
                            __bf16* __restrict__ outBf,
                            float* __restrict__ outF,
                            int M, int N, int K, int relu) {
  const int wave = threadIdx.x >> 5;
  const int lane = threadIdx.x & 31;
  const int tile = blockIdx.x * (blockDim.x >> 5) + wave;
  const int ntile64 = N >> 6;                 // 64-wide N tiles
  const int tm = tile / ntile64;              // 16-row tile index
  const int tn = (tile - tm * ntile64) * 4;   // first 16-col subtile index
  if (tm >= (M >> 4)) return;                 // wave-uniform guard

  const int half = lane >> 4;
  const int ln   = lane & 15;

  const __bf16* arow = A  + (size_t)(tm * 16 + ln) * K;
  const __bf16* brow = BT + (size_t)(tn * 16 + ln) * K;
  const size_t bstep = (size_t)16 * K;        // next 16-col subtile

  v8f acc0 = {}, acc1 = {}, acc2 = {}, acc3 = {};

  for (int kb = 0; kb < K; kb += 32) {
    union { v16bf v; v8bf h[2]; } af, bf0, bf1, bf2, bf3;
    af.h[0]  = *(const v8bf*)(arow + kb + half * 8);
    af.h[1]  = *(const v8bf*)(arow + kb + 16 + half * 8);
    bf0.h[0] = *(const v8bf*)(brow + kb + half * 8);
    bf0.h[1] = *(const v8bf*)(brow + kb + 16 + half * 8);
    bf1.h[0] = *(const v8bf*)(brow + bstep + kb + half * 8);
    bf1.h[1] = *(const v8bf*)(brow + bstep + kb + 16 + half * 8);
    bf2.h[0] = *(const v8bf*)(brow + 2 * bstep + kb + half * 8);
    bf2.h[1] = *(const v8bf*)(brow + 2 * bstep + kb + 16 + half * 8);
    bf3.h[0] = *(const v8bf*)(brow + 3 * bstep + kb + half * 8);
    bf3.h[1] = *(const v8bf*)(brow + 3 * bstep + kb + 16 + half * 8);

    acc0 = __builtin_amdgcn_wmma_f32_16x16x32_bf16(false, af.v, false, bf0.v,
                                                   (short)0, acc0, false, false);
    acc1 = __builtin_amdgcn_wmma_f32_16x16x32_bf16(false, af.v, false, bf1.v,
                                                   (short)0, acc1, false, false);
    acc2 = __builtin_amdgcn_wmma_f32_16x16x32_bf16(false, af.v, false, bf2.v,
                                                   (short)0, acc2, false, false);
    acc3 = __builtin_amdgcn_wmma_f32_16x16x32_bf16(false, af.v, false, bf3.v,
                                                   (short)0, acc3, false, false);
  }

  union { v8f v; float e[8]; } cc[4];
  cc[0].v = acc0; cc[1].v = acc1; cc[2].v = acc2; cc[3].v = acc3;

  #pragma unroll
  for (int j = 0; j < 4; ++j) {
    const int col = (tn + j) * 16 + ln;
    const float bv = bias[col];
    #pragma unroll
    for (int i = 0; i < 8; ++i) {
      const int row = tm * 16 + i + half * 8;
      float val = cc[j].e[i] + bv;
      if (relu) val = fmaxf(val, 0.0f);
      if (outBf) outBf[(size_t)row * N + col] = (__bf16)val;
      else       outF [(size_t)row * N + col] = val;
    }
  }
}

// ---------------------------------------------------------------------------
// Scores + masked softmax + entropy. One block (256 threads, 8 waves) per
// batch row b. Each wave computes one action's 512-dim dot per iteration:
// lane l owns dims [l*16, l*16+16) -> X2 slice preloaded in registers;
// lanes 0..15 read relation_emb, lanes 16..31 read entity_emb (f32, L2 hot).
// ---------------------------------------------------------------------------
__global__ void k_scores_softmax(const float* __restrict__ X2,
                                 const int* __restrict__ r_space,
                                 const int* __restrict__ e_space,
                                 const float* __restrict__ mask,
                                 const float* __restrict__ rel_emb,
                                 const float* __restrict__ ent_emb,
                                 float* __restrict__ dist_out,
                                 float* __restrict__ ent_out) {
  __shared__ float s_scores[AN];
  __shared__ float red[AN];

  const int b    = blockIdx.x;
  const int tid  = threadIdx.x;
  const int lane = tid & 31;
  const int wave = tid >> 5;

  // Preload this lane's fixed X2 slice (dims lane*16 .. lane*16+15)
  float x[16];
  const float* x2b = X2 + (size_t)b * AD + lane * 16;
  #pragma unroll
  for (int i = 0; i < 16; ++i) x[i] = x2b[i];

  for (int it = 0; it < AN / 8; ++it) {
    const int a  = wave * (AN / 8) + it;
    const int r  = r_space[(size_t)b * AN + a];
    const int ei = e_space[(size_t)b * AN + a];
    const float* ep = (lane < 16)
        ? (rel_emb + (size_t)r  * DIM + lane * 16)
        : (ent_emb + (size_t)ei * DIM + (lane - 16) * 16);
    float acc = 0.0f;
    #pragma unroll
    for (int i = 0; i < 16; ++i) acc = fmaf(ep[i], x[i], acc);
    #pragma unroll
    for (int off = 16; off > 0; off >>= 1) acc += __shfl_down(acc, off);
    if (lane == 0) {
      const float m = mask[(size_t)b * AN + a];
      s_scores[a] = acc - (1.0f - m) * HUGEF;
    }
  }
  __syncthreads();

  const float sc = s_scores[tid];

  // max
  red[tid] = sc; __syncthreads();
  for (int s = AN / 2; s > 0; s >>= 1) {
    if (tid < s) red[tid] = fmaxf(red[tid], red[tid + s]);
    __syncthreads();
  }
  const float mx = red[0]; __syncthreads();

  // exp + sum
  const float ex = expf(sc - mx);
  red[tid] = ex; __syncthreads();
  for (int s = AN / 2; s > 0; s >>= 1) {
    if (tid < s) red[tid] += red[tid + s];
    __syncthreads();
  }
  const float sum = red[0]; __syncthreads();

  const float p = ex / sum;
  dist_out[(size_t)b * AN + tid] = p;

  // entropy = -sum p * log(max(p, EPS))
  red[tid] = -p * logf(fmaxf(p, EPSF)); __syncthreads();
  for (int s = AN / 2; s > 0; s >>= 1) {
    if (tid < s) red[tid] += red[tid + s];
    __syncthreads();
  }
  if (tid == 0) ent_out[b] = red[0];
}

// ---------------------------------------------------------------------------
// Launch
// ---------------------------------------------------------------------------
extern "C" void kernel_launch(void* const* d_in, const int* in_sizes, int n_in,
                              void* d_out, int out_size, void* d_ws, size_t ws_size,
                              hipStream_t stream) {
  const int*   e        = (const int*)  d_in[0];
  const int*   q        = (const int*)  d_in[1];
  const float* hstate   = (const float*)d_in[2];
  const int*   r_space  = (const int*)  d_in[3];
  const int*   e_space  = (const int*)  d_in[4];
  const float* mask     = (const float*)d_in[5];
  const float* ent_emb  = (const float*)d_in[6];
  const float* rel_emb  = (const float*)d_in[7];
  const float* W1       = (const float*)d_in[8];   // [INDIM, AD]
  const float* b1       = (const float*)d_in[9];   // [AD]
  const float* W2       = (const float*)d_in[10];  // [AD, AD]
  const float* b2       = (const float*)d_in[11];  // [AD]

  // Workspace layout (all offsets 256B-aligned)
  char* ws = (char*)d_ws;
  const size_t OFF_W1T = 0;                                     // bf16 [AD, INDIM]
  const size_t OFF_W2T = OFF_W1T + (size_t)AD * INDIM * 2;      // bf16 [AD, AD]
  const size_t OFF_XIN = OFF_W2T + (size_t)AD * AD * 2;         // bf16 [B, INDIM]
  const size_t OFF_H   = OFF_XIN + (size_t)BATCH * INDIM * 2;   // bf16 [B, AD]
  const size_t OFF_X2  = OFF_H   + (size_t)BATCH * AD * 2;      // f32  [B, AD]
  __bf16* W1T = (__bf16*)(ws + OFF_W1T);
  __bf16* W2T = (__bf16*)(ws + OFF_W2T);
  __bf16* Xin = (__bf16*)(ws + OFF_XIN);
  __bf16* H   = (__bf16*)(ws + OFF_H);
  float*  X2  = (float*) (ws + OFF_X2);

  float* dist_out = (float*)d_out;                 // [B, AN]
  float* ent_out  = (float*)d_out + (size_t)BATCH * AN;  // [B]

  // 1) Transpose+convert weights to bf16 [N,K]
  k_transpose_bf16<<<(INDIM * AD + 255) / 256, 256, 0, stream>>>(W1, W1T, INDIM, AD);
  k_transpose_bf16<<<(AD * AD + 255) / 256, 256, 0, stream>>>(W2, W2T, AD, AD);

  // 2) Gather + concat input, convert to bf16
  k_build_xin<<<(BATCH * INDIM) / 256, 256, 0, stream>>>(e, q, hstate, ent_emb, rel_emb, Xin);

  // 3) GEMM1: H = relu(Xin @ W1 + b1)   [1024x1024x512] -> bf16
  //    tiles: (M/16)*(N/64) waves = 64*8 = 512 waves = 128 blocks of 4 waves
  {
    int waves = (BATCH / 16) * (AD / 64);
    k_wmma_gemm<<<waves / 4, 128, 0, stream>>>(Xin, W1T, b1, H, nullptr,
                                               BATCH, AD, INDIM, 1);
  }

  // 4) GEMM2: X2 = H @ W2 + b2   [1024x512x512] -> f32
  {
    int waves = (BATCH / 16) * (AD / 64);
    k_wmma_gemm<<<waves / 4, 128, 0, stream>>>(H, W2T, b2, nullptr, X2,
                                               BATCH, AD, AD, 0);
  }

  // 5) scores + softmax + entropy (one block per batch row)
  k_scores_softmax<<<BATCH, AN, 0, stream>>>(X2, r_space, e_space, mask,
                                             rel_emb, ent_emb, dist_out, ent_out);
}
